// SparseGATConv_20332375179737
// MI455X (gfx1250) — compile-verified, matrix operand order checked by reading, not run on hardware
//
#include <hip/hip_runtime.h>
#include <hip/hip_bf16.h>

#define GN 50000
#define GE 800000
#define GC 256      // IN_C and HEADS*OUT_C
#define GH 8
#define GOC 32

typedef __attribute__((ext_vector_type(16))) _Float16 v16h;
typedef __attribute__((ext_vector_type(8)))  _Float16 v8h;
typedef __attribute__((ext_vector_type(8)))  float    v8f;
typedef __attribute__((ext_vector_type(4)))  float    v4f;

// ---------------- init: out = bias broadcast, denom = 0, segmax sentinel = 0
__global__ void k_init(float* __restrict__ out, const float* __restrict__ bias,
                       unsigned* __restrict__ smax, float* __restrict__ denom) {
    int idx = blockIdx.x * blockDim.x + threadIdx.x;      // GN*GC threads
    if (idx < GN * GC) out[idx] = bias[idx & (GC - 1)];
    if (idx < GN * GH) { smax[idx] = 0u; denom[idx] = 0.0f; }
}

// ---------------- W [k][n] f32 -> WT [n][k] f16 (transposed for contiguous B-fragment loads)
__global__ void k_convW(const float* __restrict__ W, _Float16* __restrict__ WT) {
    int idx = blockIdx.x * blockDim.x + threadIdx.x;      // 256*256
    if (idx >= GC * GC) return;
    int k = idx >> 8, n = idx & 255;
    WT[n * GC + k] = (_Float16)W[idx];
}

// ---------------- h = x @ W via v_wmma_f32_16x16x32_f16, one 16x16 tile per wave
__global__ void k_gemm(const float* __restrict__ x, const _Float16* __restrict__ WT,
                       float* __restrict__ hout) {
    int g = blockIdx.x * (blockDim.x >> 5) + (threadIdx.x >> 5);  // 3125*16 tiles
    if (g >= (GN / 16) * (GC / 16)) return;
    int lane  = threadIdx.x & 31;
    int tileN = g & 15;
    int tileM = g >> 4;
    int r16 = lane & 15;
    int hf  = lane >> 4;                                  // which half of the wave

    const float*    xrow = x  + (size_t)(tileM * 16 + r16) * GC;  // A: row M = lane%16
    const _Float16* wcol = WT + (size_t)(tileN * 16 + r16) * GC;  // B: col N = lane%16

    v8f acc = {};
#pragma unroll
    for (int k0 = 0; k0 < GC; k0 += 32) {
        int kb = k0 + 8 * hf;  // 16-bit A layout: lanes 0-15 K=0..7 & 16..23, lanes 16-31 +8
        v4f x0 = *(const v4f*)(xrow + kb);
        v4f x1 = *(const v4f*)(xrow + kb + 4);
        v4f x2 = *(const v4f*)(xrow + kb + 16);
        v4f x3 = *(const v4f*)(xrow + kb + 20);
        v8h b0 = *(const v8h*)(wcol + kb);
        v8h b1 = *(const v8h*)(wcol + kb + 16);
        v16h a, b;
#pragma unroll
        for (int t = 0; t < 4; ++t) {
            a[t]      = (_Float16)x0[t];
            a[4 + t]  = (_Float16)x1[t];
            a[8 + t]  = (_Float16)x2[t];
            a[12 + t] = (_Float16)x3[t];
        }
#pragma unroll
        for (int t = 0; t < 8; ++t) { b[t] = b0[t]; b[8 + t] = b1[t]; }
        acc = __builtin_amdgcn_wmma_f32_16x16x32_f16(
            /*neg_a=*/false, a, /*neg_b=*/false, b,
            /*c_mod=*/(short)0, acc, /*reuse_a=*/false, /*reuse_b=*/false);
    }
    // C/D layout: VGPR r -> row M = tileM*16 + hf*8 + r, col N = tileN*16 + lane%16
    float* orow = hout + (size_t)(tileM * 16 + hf * 8) * GC + tileN * 16 + r16;
#pragma unroll
    for (int r = 0; r < 8; ++r) orow[r * GC] = acc[r];
}

// ---------------- a1/a2: per-(node,head) wave reduction over 32 channels
__global__ void k_attn(const float* __restrict__ h, const float* __restrict__ al,
                       const float* __restrict__ ar, float* __restrict__ a1,
                       float* __restrict__ a2) {
    int w = blockIdx.x * (blockDim.x >> 5) + (threadIdx.x >> 5);  // GN*GH waves
    if (w >= GN * GH) return;
    int lane = threadIdx.x & 31;
    int node = w >> 3, head = w & 7;
    float hv = h[(size_t)node * GC + head * GOC + lane];
    float sl = hv * al[head * GOC + lane];
    float sr = hv * ar[head * GOC + lane];
#pragma unroll
    for (int o = 16; o > 0; o >>= 1) {
        sl += __shfl_xor(sl, o, 32);
        sr += __shfl_xor(sr, o, 32);
    }
    if (lane == 0) { a1[w] = sl; a2[w] = sr; }
}

// order-preserving float -> uint encoding for atomic max
__device__ __forceinline__ unsigned f2ord(float v) {
    unsigned s = __float_as_uint(v);
    return (s & 0x80000000u) ? ~s : (s | 0x80000000u);
}

// ---------------- e = leaky_relu(a1[src]+a2[dst]); segment max via atomicMax(u32)
__global__ void k_edge(const int* __restrict__ ei, const float* __restrict__ a1,
                       const float* __restrict__ a2, float* __restrict__ eb,
                       unsigned* __restrict__ smax) {
    int idx = blockIdx.x * blockDim.x + threadIdx.x;      // GE*GH
    if (idx >= GE * GH) return;
    int e = idx >> 3, head = idx & 7;
    int src = ei[e], dst = ei[GE + e];
    float v = a1[src * GH + head] + a2[dst * GH + head];
    v = v > 0.0f ? v : 0.2f * v;
    eb[idx] = v;
    atomicMax(&smax[dst * GH + head], f2ord(v));
}

// ---------------- ex = exp(e - segmax[dst]); denom[dst] += ex (in-place on eb)
__global__ void k_exp(const int* __restrict__ ei, float* __restrict__ eb,
                      const unsigned* __restrict__ smax, float* __restrict__ denom) {
    int idx = blockIdx.x * blockDim.x + threadIdx.x;      // GE*GH
    if (idx >= GE * GH) return;
    int e = idx >> 3, head = idx & 7;
    int dst = ei[GE + e];
    unsigned u = smax[dst * GH + head];
    float m = (u == 0u) ? 0.0f
            : ((u & 0x80000000u) ? __uint_as_float(u ^ 0x80000000u) : __uint_as_float(~u));
    float ex = __expf(eb[idx] - m);
    eb[idx] = ex;
    atomicAdd(&denom[dst * GH + head], ex);
}

// ---------------- out[dst] += h[src] * alpha; one wave per edge, lane = channel%32, iter = head
__global__ void k_agg(const int* __restrict__ ei, const float* __restrict__ h,
                      const float* __restrict__ eb, const float* __restrict__ denom,
                      float* __restrict__ out) {
    int w = blockIdx.x * (blockDim.x >> 5) + (threadIdx.x >> 5);  // GE waves
    if (w >= GE) return;
    int lane = threadIdx.x & 31;
    int src = ei[w], dst = ei[GE + w];
    const float* hs = h + (size_t)src * GC;
    float* op = out + (size_t)dst * GC;
#pragma unroll
    for (int i = 0; i < GH; ++i) {
        float alpha = eb[w * GH + i] / denom[dst * GH + i];
        atomicAdd(&op[i * GOC + lane], hs[i * GOC + lane] * alpha);
    }
}

extern "C" void kernel_launch(void* const* d_in, const int* in_sizes, int n_in,
                              void* d_out, int out_size, void* d_ws, size_t ws_size,
                              hipStream_t stream) {
    const float* x    = (const float*)d_in[0];
    const int*   ei   = (const int*)d_in[1];
    const float* W    = (const float*)d_in[2];
    const float* al   = (const float*)d_in[3];
    const float* ar   = (const float*)d_in[4];
    const float* bias = (const float*)d_in[5];
    float* out = (float*)d_out;

    char* ws = (char*)d_ws;
    const size_t WT_OFF = 0;                                   // 256*256 f16  = 131072 B
    const size_t H_OFF  = WT_OFF + (size_t)GC * GC * 2;        // N*256 f32    = 51.2 MB
    const size_t A1_OFF = H_OFF + (size_t)GN * GC * 4;         // N*8 f32
    const size_t A2_OFF = A1_OFF + (size_t)GN * GH * 4;
    const size_t SM_OFF = A2_OFF + (size_t)GN * GH * 4;        // N*8 u32
    const size_t DN_OFF = SM_OFF + (size_t)GN * GH * 4;        // N*8 f32
    const size_t EB_OFF = DN_OFF + (size_t)GN * GH * 4;        // E*8 f32      = 25.6 MB

    _Float16* WT    = (_Float16*)(ws + WT_OFF);
    float*    hbuf  = (float*)(ws + H_OFF);
    float*    a1    = (float*)(ws + A1_OFF);
    float*    a2    = (float*)(ws + A2_OFF);
    unsigned* smax  = (unsigned*)(ws + SM_OFF);
    float*    denom = (float*)(ws + DN_OFF);
    float*    eb    = (float*)(ws + EB_OFF);

    // 1. init out=bias, smax=0, denom=0
    k_init<<<(GN * GC + 255) / 256, 256, 0, stream>>>(out, bias, smax, denom);
    // 2. transpose+convert W
    k_convW<<<(GC * GC + 255) / 256, 256, 0, stream>>>(W, WT);
    // 3. WMMA GEMM: 50000 tiles, 8 waves/block
    k_gemm<<<(GN / 16) * (GC / 16) / 8, 256, 0, stream>>>(x, WT, hbuf);
    // 4. attention coefficients: 400000 waves, 8 per block
    k_attn<<<(GN * GH) / 8, 256, 0, stream>>>(hbuf, al, ar, a1, a2);
    // 5. edge logits + segment max
    k_edge<<<(GE * GH + 255) / 256, 256, 0, stream>>>(ei, a1, a2, eb, smax);
    // 6. exp + denom
    k_exp<<<(GE * GH + 255) / 256, 256, 0, stream>>>(ei, eb, smax, denom);
    // 7. weighted scatter-aggregate: 800000 waves, 8 per block
    k_agg<<<GE / 8, 256, 0, stream>>>(ei, hbuf, eb, denom, out);
}
